// PointTransitionDown_32899449487855
// MI455X (gfx1250) — matrix operand order repaired and checked by reading, not compiled, not run on hardware
//
#include <hip/hip_runtime.h>

// ---------------- problem constants ----------------
#define BATCH   8
#define NPTS    8192
#define DIN     64
#define DOUT    128
#define NQRY    2048
#define NGRP    32
#define NROWS   (BATCH*NQRY*NGRP)      // 524288
#define EPSF    1e-5f
#define LDA     136                     // bf16 LDS stride (128 + 8 pad)

typedef __attribute__((ext_vector_type(16))) __bf16 v16bf;
typedef __attribute__((ext_vector_type(8)))  float  v8f;

union Frag { uint4 u[2]; v16bf v; };

// float -> bf16 (round to nearest even)
static __device__ inline unsigned short f2bf(float f) {
    unsigned int u = __float_as_uint(f);
    u += 0x7FFFu + ((u >> 16) & 1u);
    return (unsigned short)(u >> 16);
}

// ---------------- workspace layout (bytes) ----------------
#define OFF_UCE   ((size_t)0)                       // B*2048 int
#define OFF_UNE   ((size_t)65536)                   // B*2048*32 int
#define OFF_INST  (OFF_UNE + (size_t)2097152)       // f32: [0..7]sum [8..15]sq [16..23]invStd
#define OFF_STATS (OFF_INST + (size_t)512)          // 5 * 256 f32
#define OFF_SAB   (OFF_STATS + (size_t)5120)        // 5 * 256 f32
#define OFF_BUFA  ((size_t)2228224)                 // 268435456 bytes each
#define BUFSZ     ((size_t)268435456)
#define OFF_BUFB  (OFF_BUFA + BUFSZ)
#define OFF_BUFC  (OFF_BUFB + BUFSZ)

#define GEMM_SHMEM (34816 + 34816 + 512 + 512 + 2048)

// =====================================================================
// zero small stats regions
__global__ void k_zero(float* p, int n) {
    int i = blockIdx.x * blockDim.x + threadIdx.x;
    if (i < n) p[i] = 0.f;
}

// =====================================================================
// Farthest point sampling: one block per batch, 1024 threads, 8 pts/thread
__global__ __launch_bounds__(1024) void k_fps(const float* __restrict__ z,
                                              int* __restrict__ uce,
                                              float* __restrict__ zce) {
    int b = blockIdx.x, t = threadIdx.x;
    __shared__ float ls[3];
    __shared__ float wv[32];
    __shared__ int   wi[32];
    __shared__ int   sBest;
    const float* zb = z + (size_t)b * NPTS * 3;

    float px[8], py[8], pz[8], dst[8];
#pragma unroll
    for (int k = 0; k < 8; ++k) {
        int p = k * 1024 + t;
        px[k] = zb[p*3+0]; py[k] = zb[p*3+1]; pz[k] = zb[p*3+2];
        dst[k] = 1e30f;
    }
    if (t == 0) {
        ls[0] = zb[0]; ls[1] = zb[1]; ls[2] = zb[2];
        uce[b*NQRY] = 0;
        size_t o = (size_t)b*NQRY*3;
        zce[o] = zb[0]; zce[o+1] = zb[1]; zce[o+2] = zb[2];
    }
    __syncthreads();

    for (int i = 1; i < NQRY; ++i) {
        float lx = ls[0], ly = ls[1], lz = ls[2];
        float bv = -1.f; int bi = 0x7fffffff;
#pragma unroll
        for (int k = 0; k < 8; ++k) {
            float dx = px[k]-lx, dy = py[k]-ly, dz = pz[k]-lz;
            float d = dx*dx + dy*dy + dz*dz;
            dst[k] = fminf(dst[k], d);
            if (dst[k] > bv) { bv = dst[k]; bi = k*1024 + t; }
        }
        // wave32 argmax (tie -> lower index)
        for (int o = 16; o > 0; o >>= 1) {
            float ov = __shfl_down(bv, o);
            int   oi = __shfl_down(bi, o);
            if (ov > bv || (ov == bv && oi < bi)) { bv = ov; bi = oi; }
        }
        if ((t & 31) == 0) { wv[t>>5] = bv; wi[t>>5] = bi; }
        __syncthreads();
        if (t == 0) {
            float v = wv[0]; int ix = wi[0];
            for (int w = 1; w < 32; ++w)
                if (wv[w] > v || (wv[w] == v && wi[w] < ix)) { v = wv[w]; ix = wi[w]; }
            sBest = ix;
            uce[b*NQRY + i] = ix;
        }
        __syncthreads();
        int ix = sBest;
        if ((ix & 1023) == t) {
            int k = ix >> 10;
            ls[0] = px[k]; ls[1] = py[k]; ls[2] = pz[k];
            size_t o = ((size_t)b*NQRY + i) * 3;
            zce[o] = px[k]; zce[o+1] = py[k]; zce[o+2] = pz[k];
        }
        __syncthreads();
    }
}

// =====================================================================
// 32-NN per query: one block (128 thr) per query; dists in LDS; 32 argmin rounds
__global__ __launch_bounds__(128) void k_knn(const float* __restrict__ z,
                                             const float* __restrict__ zce,
                                             int* __restrict__ une) {
    int q = blockIdx.x, t = threadIdx.x;
    int b = q >> 11;
    __shared__ float dl[NPTS];
    __shared__ float wv[4];
    __shared__ int   wi[4];
    const float* zb = z + (size_t)b * NPTS * 3;
    float qx = zce[(size_t)q*3], qy = zce[(size_t)q*3+1], qz = zce[(size_t)q*3+2];
#pragma unroll 4
    for (int j = 0; j < 64; ++j) {
        int p = j*128 + t;
        float dx = zb[p*3]-qx, dy = zb[p*3+1]-qy, dz = zb[p*3+2]-qz;
        dl[p] = dx*dx + dy*dy + dz*dz;
    }
    __syncthreads();
    for (int r = 0; r < NGRP; ++r) {
        float bv = 1e30f; int bi = 0x7fffffff;
        for (int j = 0; j < 64; ++j) {
            int p = j*128 + t;
            float v = dl[p];
            if (v < bv) { bv = v; bi = p; }
        }
        for (int o = 16; o > 0; o >>= 1) {
            float ov = __shfl_down(bv, o);
            int   oi = __shfl_down(bi, o);
            if (ov < bv || (ov == bv && oi < bi)) { bv = ov; bi = oi; }
        }
        if ((t & 31) == 0) { wv[t>>5] = bv; wi[t>>5] = bi; }
        __syncthreads();
        if (t == 0) {
            float v = wv[0]; int ix = wi[0];
            for (int w = 1; w < 4; ++w)
                if (wv[w] < v || (wv[w] == v && wi[w] < ix)) { v = wv[w]; ix = wi[w]; }
            une[(size_t)q*NGRP + r] = ix;
            dl[ix] = 1e30f;
        }
        __syncthreads();
    }
}

// =====================================================================
// per-batch std of (x_ne - x_mu): sum & sumsq via atomics
__global__ __launch_bounds__(256) void k_instats(const float* __restrict__ x,
                                                 const int* __restrict__ une,
                                                 const int* __restrict__ uce,
                                                 float* __restrict__ acc) {
    __shared__ float rs[256], rq[256];
    int t = threadIdx.x;
    int rowBase = blockIdx.x * 128;
    int row = t >> 1, half = t & 1;
    int grow = rowBase + row;
    int b = grow >> 16;
    int in = une[grow];
    int ic = uce[grow >> 5];
    const float* xn = x + ((size_t)b*NPTS + in) * DIN;
    const float* xc = x + ((size_t)b*NPTS + ic) * DIN;
    float s = 0.f, qq = 0.f;
    int c0 = half * 32;
#pragma unroll
    for (int c = 0; c < 32; ++c) {
        float d = xn[c0+c] - xc[c0+c];
        s += d; qq += d*d;
    }
    rs[t] = s; rq[t] = qq;
    __syncthreads();
    for (int o = 128; o > 0; o >>= 1) {
        if (t < o) { rs[t] += rs[t+o]; rq[t] += rq[t+o]; }
        __syncthreads();
    }
    if (t == 0) { atomicAdd(&acc[b], rs[0]); atomicAdd(&acc[8+b], rq[0]); }
}

__global__ void k_infin(float* acc) {
    int t = threadIdx.x;
    if (t < 8) {
        float n = (float)(NQRY * NGRP * DIN);      // 4194304
        float s = acc[t], sq = acc[8+t];
        float var = (sq - s*s/n) / (n - 1.f);
        float sd = sqrtf(fmaxf(var, 0.f));
        acc[16+t] = 1.f / (sd + EPSF);
    }
}

// =====================================================================
// BN finalize: stats -> per-channel (scale,bias)
__global__ void k_bnfin(const float* __restrict__ stats,
                        const float* __restrict__ g,
                        const float* __restrict__ bb,
                        float* __restrict__ sab) {
    int c = threadIdx.x;
    const float n = (float)NROWS;
    float mean = stats[c] / n;
    float var = stats[128+c] / n - mean*mean;
    float sc = g[c] * rsqrtf(var + EPSF);
    sab[c] = sc;
    sab[128+c] = bb[c] - mean * sc;
}

// =====================================================================
// shared GEMM core: wave computes 16x128 strip, C = A(128x128 bf16) @ Wt
static __device__ inline void load_wt(const float* __restrict__ W, unsigned short* Wl) {
    int t = threadIdx.x;
#pragma unroll 4
    for (int i = 0; i < 64; ++i) {
        int idx = t + i*256;
        int k = idx >> 7, n = idx & 127;
        Wl[n*LDA + k] = f2bf(W[idx]);
    }
}

static __device__ inline void gemm_core(const unsigned short* __restrict__ Al,
                                        const unsigned short* __restrict__ Wl,
                                        float* sSum, float* sSq,
                                        float* __restrict__ rawOut, int rowBase) {
    int t = threadIdx.x;
    int lane = t & 31, wid = t >> 5;
    int m0 = wid * 16;
    int khalf = lane >> 4;
    int rsub = lane & 15;

    Frag a[4];
#pragma unroll
    for (int kb = 0; kb < 4; ++kb) {
        const unsigned short* ap = Al + (m0 + rsub)*LDA + kb*32 + khalf*8;
        a[kb].u[0] = *(const uint4*)ap;          // K = kb*32 + 8*khalf + [0..7]
        a[kb].u[1] = *(const uint4*)(ap + 16);   // K = kb*32 + 16 + 8*khalf + [0..7]
    }
#pragma unroll
    for (int nt = 0; nt < 8; ++nt) {
        int n0 = nt * 16;
        v8f acc = {};
#pragma unroll
        for (int kb = 0; kb < 4; ++kb) {
            Frag bfr;
            const unsigned short* bp = Wl + (n0 + rsub)*LDA + kb*32 + khalf*8;
            bfr.u[0] = *(const uint4*)bp;        // K = kb*32 + 8*khalf + [0..7]
            bfr.u[1] = *(const uint4*)(bp + 16); // K = kb*32 + 16 + 8*khalf + [0..7]
            acc = __builtin_amdgcn_wmma_f32_16x16x32_bf16(
                false, a[kb].v, false, bfr.v, (short)0, acc, false, false);
        }
        int col = n0 + rsub;
        float s = 0.f, sq = 0.f;
        size_t gr0 = (size_t)(rowBase + m0 + khalf*8);
#pragma unroll
        for (int r = 0; r < 8; ++r) {
            float v = acc[r];
            rawOut[(gr0 + r)*DOUT + col] = v;
            s += v; sq += v*v;
        }
        atomicAdd(&sSum[col], s);
        atomicAdd(&sSq[col], sq);
    }
}

// ---- layer 0: build [norm_diff | x_mu] features on the fly, @W0 ----
__global__ __launch_bounds__(256) void k_gemm_feat(const float* __restrict__ x,
                                                   const int* __restrict__ une,
                                                   const int* __restrict__ uce,
                                                   const float* __restrict__ alpha,
                                                   const float* __restrict__ beta,
                                                   const float* __restrict__ inst,
                                                   const float* __restrict__ W,
                                                   float* __restrict__ rawOut,
                                                   float* __restrict__ stats) {
    extern __shared__ unsigned char smem[];
    unsigned short* Al = (unsigned short*)smem;
    unsigned short* Wl = Al + 128*LDA;
    float* sSum = (float*)(Wl + 128*LDA);
    float* sSq  = sSum + 128;
    int t = threadIdx.x;
    int rowBase = blockIdx.x * 128;
    if (t < 128) { sSum[t] = 0.f; sSq[t] = 0.f; }
    load_wt(W, Wl);

    int row = t >> 1, half = t & 1;
    int grow = rowBase + row;
    int b = grow >> 16;
    int in = une[grow];
    int ic = uce[grow >> 5];
    const float* xn = x + ((size_t)b*NPTS + in) * DIN;
    const float* xc = x + ((size_t)b*NPTS + ic) * DIN;
    if (half == 0) {
        float is = inst[16 + b];
#pragma unroll 4
        for (int c = 0; c < 64; ++c)
            Al[row*LDA + c] = f2bf(alpha[c] * (xn[c] - xc[c]) * is + beta[c]);
    } else {
#pragma unroll 4
        for (int c = 0; c < 64; ++c)
            Al[row*LDA + 64 + c] = f2bf(xc[c]);
    }
    __syncthreads();
    gemm_core(Al, Wl, sSum, sSq, rawOut, rowBase);
    __syncthreads();
    if (t < 128) { atomicAdd(&stats[t], sSum[t]); atomicAdd(&stats[128+t], sSq[t]); }
}

// ---- layers 1,2,4: A = relu(raw*scale+bias), @W ----
__global__ __launch_bounds__(256) void k_gemm_bnrelu(const float* __restrict__ rawIn,
                                                     const float* __restrict__ sab,
                                                     const float* __restrict__ W,
                                                     float* __restrict__ rawOut,
                                                     float* __restrict__ stats) {
    extern __shared__ unsigned char smem[];
    unsigned short* Al = (unsigned short*)smem;
    unsigned short* Wl = Al + 128*LDA;
    float* sSum = (float*)(Wl + 128*LDA);
    float* sSq  = sSum + 128;
    float* sA   = sSq + 128;
    float* sB   = sA + 128;
    int t = threadIdx.x;
    int rowBase = blockIdx.x * 128;
    if (t < 128) { sSum[t] = 0.f; sSq[t] = 0.f; sA[t] = sab[t]; sB[t] = sab[128+t]; }
    load_wt(W, Wl);
    __syncthreads();

    int row = t >> 1, half = t & 1;
    int c0 = half * 64;
    const float4* rp = (const float4*)(rawIn + ((size_t)(rowBase+row))*DOUT + c0);
#pragma unroll 4
    for (int i = 0; i < 16; ++i) {
        float4 f = rp[i];
        int ch = c0 + i*4;
        Al[row*LDA + ch + 0] = f2bf(fmaxf(f.x*sA[ch+0] + sB[ch+0], 0.f));
        Al[row*LDA + ch + 1] = f2bf(fmaxf(f.y*sA[ch+1] + sB[ch+1], 0.f));
        Al[row*LDA + ch + 2] = f2bf(fmaxf(f.z*sA[ch+2] + sB[ch+2], 0.f));
        Al[row*LDA + ch + 3] = f2bf(fmaxf(f.w*sA[ch+3] + sB[ch+3], 0.f));
    }
    __syncthreads();
    gemm_core(Al, Wl, sSum, sSq, rawOut, rowBase);
    __syncthreads();
    if (t < 128) { atomicAdd(&stats[t], sSum[t]); atomicAdd(&stats[128+t], sSq[t]); }
}

// ---- layer 3: A = h1 = relu(bn(raw2) + relu(bn(raw0))); h1 stored back; @W2a ----
__global__ __launch_bounds__(256) void k_gemm_resid(const float* __restrict__ raw2,
                                                    const float* __restrict__ sab2,
                                                    float* __restrict__ h1buf,   // raw0 in, h1 out
                                                    const float* __restrict__ sab0,
                                                    const float* __restrict__ W,
                                                    float* __restrict__ rawOut,
                                                    float* __restrict__ stats) {
    extern __shared__ unsigned char smem[];
    unsigned short* Al = (unsigned short*)smem;
    unsigned short* Wl = Al + 128*LDA;
    float* sSum = (float*)(Wl + 128*LDA);
    float* sSq  = sSum + 128;
    float* sA2  = sSq + 128;
    float* sB2  = sA2 + 128;
    float* sA0  = sB2 + 128;
    float* sB0  = sA0 + 128;
    int t = threadIdx.x;
    int rowBase = blockIdx.x * 128;
    if (t < 128) {
        sSum[t] = 0.f; sSq[t] = 0.f;
        sA2[t] = sab2[t]; sB2[t] = sab2[128+t];
        sA0[t] = sab0[t]; sB0[t] = sab0[128+t];
    }
    load_wt(W, Wl);
    __syncthreads();

    int row = t >> 1, half = t & 1;
    int c0 = half * 64;
    size_t base = ((size_t)(rowBase+row))*DOUT + c0;
    const float4* r2 = (const float4*)(raw2 + base);
    float4*       r0 = (float4*)(h1buf + base);
#pragma unroll 4
    for (int i = 0; i < 16; ++i) {
        float4 f2 = r2[i];
        float4 f0 = r0[i];
        int ch = c0 + i*4;
        float4 h;
        h.x = fmaxf(f2.x*sA2[ch+0]+sB2[ch+0] + fmaxf(f0.x*sA0[ch+0]+sB0[ch+0], 0.f), 0.f);
        h.y = fmaxf(f2.y*sA2[ch+1]+sB2[ch+1] + fmaxf(f0.y*sA0[ch+1]+sB0[ch+1], 0.f), 0.f);
        h.z = fmaxf(f2.z*sA2[ch+2]+sB2[ch+2] + fmaxf(f0.z*sA0[ch+2]+sB0[ch+2], 0.f), 0.f);
        h.w = fmaxf(f2.w*sA2[ch+3]+sB2[ch+3] + fmaxf(f0.w*sA0[ch+3]+sB0[ch+3], 0.f), 0.f);
        r0[i] = h;  // persist h1 for final residual
        Al[row*LDA + ch + 0] = f2bf(h.x);
        Al[row*LDA + ch + 1] = f2bf(h.y);
        Al[row*LDA + ch + 2] = f2bf(h.z);
        Al[row*LDA + ch + 3] = f2bf(h.w);
    }
    __syncthreads();
    gemm_core(Al, Wl, sSum, sSq, rawOut, rowBase);
    __syncthreads();
    if (t < 128) { atomicAdd(&stats[t], sSum[t]); atomicAdd(&stats[128+t], sSq[t]); }
}

// ---- final: h2 = relu(bn(raw4) + h1); x_out = max over group axis ----
__global__ __launch_bounds__(128) void k_final(const float* __restrict__ raw4,
                                               const float* __restrict__ sab4,
                                               const float* __restrict__ h1,
                                               float* __restrict__ out) {
    int q = blockIdx.x, c = threadIdx.x;
    float sA = sab4[c], sB = sab4[128+c];
    float m = -1e30f;
    size_t base = (size_t)q * NGRP * DOUT + c;
#pragma unroll 4
    for (int g = 0; g < NGRP; ++g) {
        size_t o = base + (size_t)g*DOUT;
        float v = fmaxf(raw4[o]*sA + sB + h1[o], 0.f);
        m = fmaxf(m, v);
    }
    out[(size_t)q*DOUT + c] = m;
}

// =====================================================================
extern "C" void kernel_launch(void* const* d_in, const int* in_sizes, int n_in,
                              void* d_out, int out_size, void* d_ws, size_t ws_size,
                              hipStream_t stream) {
    const float* x     = (const float*)d_in[0];
    const float* z     = (const float*)d_in[1];
    const float* alpha = (const float*)d_in[2];
    const float* beta  = (const float*)d_in[3];
    const float* W0    = (const float*)d_in[4];
    const float* g0    = (const float*)d_in[5];
    const float* b0    = (const float*)d_in[6];
    const float* W1a   = (const float*)d_in[7];
    const float* g1a   = (const float*)d_in[8];
    const float* b1a   = (const float*)d_in[9];
    const float* W1b   = (const float*)d_in[10];
    const float* g1b   = (const float*)d_in[11];
    const float* b1b   = (const float*)d_in[12];
    const float* W2a   = (const float*)d_in[13];
    const float* g2a   = (const float*)d_in[14];
    const float* b2a   = (const float*)d_in[15];
    const float* W2b   = (const float*)d_in[16];
    const float* g2b   = (const float*)d_in[17];
    const float* b2b   = (const float*)d_in[18];

    float* xout = (float*)d_out;                         // (B,2048,128)
    float* zce  = (float*)d_out + (size_t)BATCH*NQRY*DOUT; // (B,2048,3)

    unsigned char* ws = (unsigned char*)d_ws;
    int*   uce   = (int*)(ws + OFF_UCE);
    int*   une   = (int*)(ws + OFF_UNE);
    float* inst  = (float*)(ws + OFF_INST);
    float* stats = (float*)(ws + OFF_STATS);
    float* sab   = (float*)(ws + OFF_SAB);
    float* bufA  = (float*)(ws + OFF_BUFA);
    float* bufB  = (float*)(ws + OFF_BUFB);
    float* bufC  = (float*)(ws + OFF_BUFC);

    const int GEMM_GRID = NROWS / 128;  // 4096

    // zero inst (24 f) + stats (5*256 f): zero 1408 floats from OFF_INST
    k_zero<<<6, 256, 0, stream>>>(inst, 1408);

    k_fps<<<BATCH, 1024, 0, stream>>>(z, uce, zce);
    k_knn<<<BATCH*NQRY, 128, 0, stream>>>(z, zce, une);

    k_instats<<<GEMM_GRID, 256, 0, stream>>>(x, une, uce, inst);
    k_infin<<<1, 32, 0, stream>>>(inst);

    // layer 0
    k_gemm_feat<<<GEMM_GRID, 256, GEMM_SHMEM, stream>>>(x, une, uce, alpha, beta,
                                                        inst, W0, bufA, stats + 0*256);
    k_bnfin<<<1, 128, 0, stream>>>(stats + 0*256, g0, b0, sab + 0*256);
    // layer 1: h0 = relu(bn0(raw0)); raw1 = h0 @ W1a
    k_gemm_bnrelu<<<GEMM_GRID, 256, GEMM_SHMEM, stream>>>(bufA, sab + 0*256, W1a,
                                                          bufB, stats + 1*256);
    k_bnfin<<<1, 128, 0, stream>>>(stats + 1*256, g1a, b1a, sab + 1*256);
    // layer 2: t = relu(bn1a(raw1)); raw2 = t @ W1b
    k_gemm_bnrelu<<<GEMM_GRID, 256, GEMM_SHMEM, stream>>>(bufB, sab + 1*256, W1b,
                                                          bufC, stats + 2*256);
    k_bnfin<<<1, 128, 0, stream>>>(stats + 2*256, g1b, b1b, sab + 2*256);
    // layer 3: h1 = relu(bn1b(raw2) + h0) (stored over bufA); raw3 = h1 @ W2a
    k_gemm_resid<<<GEMM_GRID, 256, GEMM_SHMEM, stream>>>(bufC, sab + 2*256, bufA,
                                                         sab + 0*256, W2a, bufB,
                                                         stats + 3*256);
    k_bnfin<<<1, 128, 0, stream>>>(stats + 3*256, g2a, b2a, sab + 3*256);
    // layer 4: t = relu(bn2a(raw3)); raw4 = t @ W2b
    k_gemm_bnrelu<<<GEMM_GRID, 256, GEMM_SHMEM, stream>>>(bufB, sab + 3*256, W2b,
                                                          bufC, stats + 4*256);
    k_bnfin<<<1, 128, 0, stream>>>(stats + 4*256, g2b, b2b, sab + 4*256);
    // final: h2 = relu(bn2b(raw4) + h1); max over group
    k_final<<<BATCH*NQRY, 128, 0, stream>>>(bufC, sab + 4*256, bufA, xout);

    (void)in_sizes; (void)n_in; (void)out_size; (void)ws_size;
}
